// Quantizer_40853728919862
// MI455X (gfx1250) — matrix operand order repaired and checked by reading, not compiled, not run on hardware
//
#include <hip/hip_runtime.h>
#include <hip/hip_bf16.h>

typedef __attribute__((ext_vector_type(16))) _Float16 v16h;
typedef __attribute__((ext_vector_type(8)))  _Float16 v8h;
typedef __attribute__((ext_vector_type(8)))  float    v8f;

typedef int v4i __attribute__((vector_size(16)));
typedef __attribute__((address_space(1))) v4i as1_v4i;
typedef __attribute__((address_space(3))) v4i as3_v4i;

#define L_LAT   8
#define K_CODES 512
#define C_DIM   64
#define HW      1024      // 32*32
#define N_IMG   32
#define K_CH    64        // codes per LDS chunk (double buffered)
#define NCHUNK  (K_CODES / K_CH)   // 8
#define EPAD    72        // halves stride per code row in LDS (16B aligned, bank-spread)
#define WAVES   8
#define ROWS_PER_WG (WAVES*16)  // 128 spatial positions per workgroup

#if __has_builtin(__builtin_amdgcn_global_load_async_to_lds_b128)
#define HAVE_ASYNC_LDS 1
#else
#define HAVE_ASYNC_LDS 0
#endif

__device__ __forceinline__ as1_v4i* to_glob16(const void* p) {
  return (as1_v4i*)(uintptr_t)p;           // global VA is identical in AS1
}
__device__ __forceinline__ as3_v4i* to_lds16(void* p) {
  // generic pointers to LDS carry the LDS byte offset in the low 32 bits
  return (as3_v4i*)(uint32_t)(uintptr_t)p;
}

// ---------------- prep: split e into f16 hi/lo + fp32 norms ----------------
__global__ void eprep_kernel(const float* __restrict__ e,
                             _Float16* __restrict__ ehi,
                             _Float16* __restrict__ elo,
                             float* __restrict__ esq) {
  int tid = blockIdx.x * 256 + threadIdx.x;        // 262144 = 8*512*64 threads
  float x = e[tid];
  _Float16 h = (_Float16)x;
  ehi[tid] = h;
  elo[tid] = (_Float16)(x - (float)h);
  if (tid < L_LAT * K_CODES) {                     // per-code squared norm
    const float* row = e + (size_t)tid * C_DIM;
    float s = 0.f;
#pragma unroll
    for (int c = 0; c < C_DIM; ++c) s += row[c] * row[c];
    esq[tid] = s;
  }
}

// ---------------- main: WMMA cross-term + argmin + gather ----------------
__global__ void __launch_bounds__(256)
vq_main_kernel(const float* __restrict__ z, const float* __restrict__ e,
               const _Float16* __restrict__ ehi, const _Float16* __restrict__ elo,
               const float* __restrict__ esq,
               float* __restrict__ zq, float* __restrict__ idxout) {
  // [buffer][hi=0/lo=1][code*EPAD + c] : 2*2*64*72*2B = 36 KB
  __shared__ _Float16 s_e[2][2][K_CH * EPAD];
  __shared__ int      s_idx[WAVES][16];

  const int bid = blockIdx.x;               // 2048 blocks
  const int l   = bid >> 8;                 // [0,8)
  const int rem = bid & 255;
  const int n   = rem >> 3;                 // [0,32)
  const int qc  = rem & 7;                  // [0,8) chunk of 128 hw positions
  const int hw_base = qc * ROWS_PER_WG;

  const int lane = threadIdx.x & 31;
  const int wave = threadIdx.x >> 5;
  const int mcol = lane & 15;               // M row (A) / N col (B,C)
  const int hh   = lane >> 4;               // lane half

  // ---- load this wave's 16-row A tile (rows = consecutive hw), split hi/lo ----
  // ISA A layout (16-bit 16x32): lane m = lane%16; element j -> K = (j%8)+16*(j/8)+8*(lane/16)
  const int row = hw_base + wave * 16 + mcol;
  const float* zbase = z + (size_t)n * (512u * HW) + (size_t)l * (C_DIM * HW) + row;
  v16h a_hi[2], a_lo[2];
#pragma unroll
  for (int kh = 0; kh < 2; ++kh) {
#pragma unroll
    for (int j = 0; j < 16; ++j) {
      int c = kh * 32 + (j & 7) + 16 * (j >> 3) + 8 * hh;
      float x = zbase[(size_t)c * HW];      // 16 consecutive rows across lanes -> 64B segments
      _Float16 hi = (_Float16)x;
      a_hi[kh][j] = hi;
      a_lo[kh][j] = (_Float16)(x - (float)hi);
    }
  }

  // ---- per-wave chunk fill: 4 x b128 async DMA per wave (1024 16B-chunks / WG) ----
  const _Float16* gsrc[2] = {ehi, elo};
  auto fill_chunk = [&](int buf, int chunk) {
    const int cbase = (l * K_CODES + chunk * K_CH) * C_DIM;   // halves
#pragma unroll
    for (int k = 0; k < 4; ++k) {
      int cid = (wave * 4 + k) * 32 + lane;   // [0,1024)
      int arr = cid >> 9;                     // 0=hi, 1=lo
      int c2  = cid & 511;
      int r   = c2 >> 3;                      // code row [0,64)
      int col = c2 & 7;                       // 16B chunk within row
      const _Float16* src = gsrc[arr] + cbase + r * C_DIM + col * 8;
      _Float16* dst = &s_e[buf][arr][r * EPAD + col * 8];
#if HAVE_ASYNC_LDS
      __builtin_amdgcn_global_load_async_to_lds_b128(to_glob16(src), to_lds16(dst), 0, 0);
#else
      *(v8h*)dst = *(const v8h*)src;
#endif
    }
  };

  float bestv[8];
  int   besti[8];
#pragma unroll
  for (int j = 0; j < 8; ++j) { bestv[j] = 3.4e38f; besti[j] = 0; }

  fill_chunk(0, 0);                          // prologue fill of buffer 0

  for (int chunk = 0; chunk < NCHUNK; ++chunk) {
    const int cur = chunk & 1;
    __syncthreads();                         // everyone done reading buf[cur^1]
    if (chunk + 1 < NCHUNK) fill_chunk(cur ^ 1, chunk + 1);   // overlap next fill
#if HAVE_ASYNC_LDS
    if (chunk + 1 < NCHUNK) {
      // own fill(chunk) ops (4, issued earlier) are done when <=4 remain (in-order)
#if __has_builtin(__builtin_amdgcn_s_wait_asynccnt)
      __builtin_amdgcn_s_wait_asynccnt(4);
#else
      asm volatile("s_wait_asynccnt 0x4" ::: "memory");
#endif
    } else {
#if __has_builtin(__builtin_amdgcn_s_wait_asynccnt)
      __builtin_amdgcn_s_wait_asynccnt(0);
#else
      asm volatile("s_wait_asynccnt 0x0" ::: "memory");
#endif
    }
#endif
    __syncthreads();                         // all waves' fill(chunk) visible in LDS

    const _Float16* sh = s_e[cur][0];
    const _Float16* sl = s_e[cur][1];
    for (int ct = 0; ct < K_CH / 16; ++ct) { // 4 column tiles of 16 codes
      v8f C = {};
#pragma unroll
      for (int kh = 0; kh < 2; ++kh) {
        // ISA B layout (16-bit 32x16): col = lane%16; element j -> K = (j%8)+16*(j/8)+8*(lane/16)
        int boff = (ct * 16 + mcol) * EPAD + kh * 32 + 8 * hh;   // 16B-aligned
        union { v16h v; v8h p[2]; } bh, bl;
        const v8h* ph = (const v8h*)(sh + boff);
        const v8h* pl = (const v8h*)(sl + boff);
        bh.p[0] = ph[0]; bh.p[1] = ph[2];
        bl.p[0] = pl[0]; bl.p[1] = pl[2];
        // cross ~= zh*eh + zl*eh + zh*el   (fp32 accumulate; ~2^-22 rel err)
        C = __builtin_amdgcn_wmma_f32_16x16x32_f16(false, a_hi[kh], false, bh.v,
                                                   (short)0, C, false, false);
        C = __builtin_amdgcn_wmma_f32_16x16x32_f16(false, a_lo[kh], false, bh.v,
                                                   (short)0, C, false, false);
        C = __builtin_amdgcn_wmma_f32_16x16x32_f16(false, a_hi[kh], false, bl.v,
                                                   (short)0, C, false, false);
      }
      int  kidx = chunk * K_CH + ct * 16 + mcol;
      float eq  = esq[l * K_CODES + kidx];   // L2-resident, 4B/lane
#pragma unroll
      for (int j = 0; j < 8; ++j) {
        float d = eq - 2.0f * C[j];          // z_sq dropped (argmin-invariant)
        if (d < bestv[j]) { bestv[j] = d; besti[j] = kidx; }  // keeps first occurrence
      }
    }
  }

  // ---- cross-lane argmin merge within each 16-lane half (first-index ties) ----
#pragma unroll
  for (int j = 0; j < 8; ++j) {
    float v = bestv[j]; int ix = besti[j];
    for (int m = 1; m < 16; m <<= 1) {
      float ov = __shfl_xor(v, m, 32);
      int   oi = __shfl_xor(ix, m, 32);
      if (ov < v || (ov == v && oi < ix)) { v = ov; ix = oi; }
    }
    bestv[j] = v; besti[j] = ix;
  }
  if (lane == 0) {
#pragma unroll
    for (int j = 0; j < 8; ++j) s_idx[wave][j] = besti[j];       // rows 0..7
  }
  if (lane == 16) {
#pragma unroll
    for (int j = 0; j < 8; ++j) s_idx[wave][8 + j] = besti[j];   // rows 8..15
  }
  __syncthreads();

  // ---- index output (as float; values are small exact ints) ----
  if (lane < 16) {
    int hw = hw_base + wave * 16 + lane;
    idxout[(size_t)l * (N_IMG * HW) + (size_t)n * HW + hw] = (float)s_idx[wave][lane];
  }

  // ---- z_q gather + coalesced store: lane r=lane%16 -> row, lane/16 -> channel parity
  {
    int r    = lane & 15;
    int code = s_idx[wave][r];
    const float* erow = e + ((size_t)l * K_CODES + code) * C_DIM;
    int hw = hw_base + wave * 16 + r;
    size_t obase = (size_t)n * (512u * HW) + (size_t)l * (C_DIM * HW) + hw;
#pragma unroll 4
    for (int i2 = 0; i2 < 32; ++i2) {
      int c = i2 * 2 + hh;
      zq[obase + (size_t)c * HW] = erow[c];  // 16 consecutive rows/lane-group -> 64B stores
    }
  }
}

extern "C" void kernel_launch(void* const* d_in, const int* in_sizes, int n_in,
                              void* d_out, int out_size, void* d_ws, size_t ws_size,
                              hipStream_t stream) {
  (void)in_sizes; (void)n_in; (void)out_size; (void)ws_size;
  const float* z = (const float*)d_in[0];   // [32,512,32,32]
  const float* e = (const float*)d_in[1];   // [8,512,64]

  _Float16* ehi = (_Float16*)d_ws;                 // 262144 halves
  _Float16* elo = ehi + (size_t)L_LAT * K_CODES * C_DIM;
  float*    esq = (float*)(elo + (size_t)L_LAT * K_CODES * C_DIM);  // 4096 f32

  float* zq     = (float*)d_out;                   // 16,777,216 floats
  float* idxout = zq + (size_t)N_IMG * 512 * HW;   // 262,144 floats

  eprep_kernel<<<dim3((L_LAT * K_CODES * C_DIM) / 256), dim3(256), 0, stream>>>(e, ehi, elo, esq);
  vq_main_kernel<<<dim3(L_LAT * N_IMG * (HW / ROWS_PER_WG)), dim3(256), 0, stream>>>(
      z, e, ehi, elo, esq, zq, idxout);
}
